// M2RNNAttn_52578989637693
// MI455X (gfx1250) — matrix operand order repaired and checked by reading, not compiled, not run on hardware
//
#include <hip/hip_runtime.h>
#include <stdint.h>

// ---------------------------------------------------------------------------
// Types for CDNA5 WMMA (wave32): v_wmma_f32_16x16x32_bf16
// ---------------------------------------------------------------------------
typedef __attribute__((ext_vector_type(16))) __bf16 v16bf;
typedef __attribute__((ext_vector_type(8)))  float  v8f;

union FragU {
    uint4 q[2];
    v16bf v;
};

__device__ inline v8f v8f_zero() {
    v8f z;
#pragma unroll
    for (int i = 0; i < 8; ++i) z[i] = 0.0f;
    return z;
}

// native f32 -> bf16 (compiler lowers to hardware cvt on gfx1250)
__device__ inline uint16_t f2bf(float f) {
    __bf16 h = (__bf16)f;
    return __builtin_bit_cast(uint16_t, h);
}

// packed pair convert: one v_cvt_pk_bf16_f32 when available
#if defined(__gfx1250__) && __has_builtin(__builtin_amdgcn_cvt_pk_bf16_f32)
typedef __attribute__((ext_vector_type(2))) __bf16 v2bf;
__device__ inline uint32_t pack2bf(float a, float b) {
    v2bf p = __builtin_amdgcn_cvt_pk_bf16_f32(a, b);
    return __builtin_bit_cast(uint32_t, p);
}
#else
__device__ inline uint32_t pack2bf(float a, float b) {
    return (uint32_t)f2bf(a) | ((uint32_t)f2bf(b) << 16);
}
#endif

// fast tanh on the native transcendental pipe: v_exp_f32 + v_rcp_f32
__device__ inline float fast_tanh(float x) {
    float ax = __builtin_fabsf(x);
    float e  = __expf(-2.0f * ax);
    float t  = (1.0f - e) * __builtin_amdgcn_rcpf(1.0f + e);
    return __builtin_copysignf(t, x);
}

// LDS byte offset of a __shared__ object (generic ptr low 32 bits = LDS addr)
__device__ inline uint32_t lds_off(const void* p) {
    return (uint32_t)(uintptr_t)p;
}

// async global->LDS copy, 16 bytes per lane (tracked by ASYNCcnt)
__device__ inline void async_copy_b128(uint32_t lds_addr, const void* sbase, uint32_t voff) {
    asm volatile("global_load_async_to_lds_b128 %0, %1, %2"
                 :: "v"(lds_addr), "v"(voff), "s"(sbase) : "memory");
}
__device__ inline void wait_async0() {
    asm volatile("s_wait_asynccnt 0" ::: "memory");
}

// Build A-fragment from 16 fp32 in registers (per ISA 16-bit A 16x32 layout):
// u32[j] packs K={base+2j, base+2j+1}; first 4 dwords K base, next 4 K base+16.
__device__ inline v16bf build_fragA(float4 f0, float4 f1, float4 f2, float4 f3) {
    FragU u;
    u.q[0].x = pack2bf(f0.x, f0.y); u.q[0].y = pack2bf(f0.z, f0.w);
    u.q[0].z = pack2bf(f1.x, f1.y); u.q[0].w = pack2bf(f1.z, f1.w);
    u.q[1].x = pack2bf(f2.x, f2.y); u.q[1].y = pack2bf(f2.z, f2.w);
    u.q[1].z = pack2bf(f3.x, f3.y); u.q[1].w = pack2bf(f3.z, f3.w);
    return u.v;
}

// A-fragment (16x32 bf16) from LDS row-major storage: two ds_load_b128.
__device__ inline v16bf load_fragA(const uint16_t* base, int ld, int mbase, int kb, int lane) {
    int half = lane >> 4, r = lane & 15;
    const uint16_t* row = base + (mbase + r) * ld;
    FragU u;
    u.q[0] = *(const uint4*)(row + kb + half * 8);
    u.q[1] = *(const uint4*)(row + kb + 16 + half * 8);
    return u.v;
}
// B-fragment (32x16 bf16) from TRANSPOSED storage [n][k]: two ds_load_b128.
__device__ inline v16bf load_fragB(const uint16_t* baseT, int ld, int nbase, int kb, int lane) {
    int half = lane >> 4, c = lane & 15;
    const uint16_t* row = baseT + (nbase + c) * ld;
    FragU u;
    u.q[0] = *(const uint4*)(row + kb + half * 8);
    u.q[1] = *(const uint4*)(row + kb + 16 + half * 8);
    return u.v;
}

// ---------------------------------------------------------------------------
// Weight packing kernels (fp32 -> bf16, transposed for B-fragment layout)
// ---------------------------------------------------------------------------
__global__ __launch_bounds__(256) void pack_qkv_w(const float* __restrict__ Wq,
                                                  const float* __restrict__ Wk,
                                                  const float* __restrict__ Wv,
                                                  uint16_t* __restrict__ Wt) {
    int idx = blockIdx.x * 256 + threadIdx.x;  // 1536*1024
    if (idx >= 1536 * 1024) return;
    int n = idx >> 10, d = idx & 1023;
    const float* src = (n < 512) ? Wq : ((n < 1024) ? Wk : Wv);
    Wt[idx] = f2bf(src[d * 512 + (n & 511)]);  // Wt[n][d] = W[d][n%512]
}

__global__ __launch_bounds__(256) void pack_wo(const float* __restrict__ Wo,
                                               uint16_t* __restrict__ WoT) {
    int idx = blockIdx.x * 256 + threadIdx.x;  // 1024*512
    if (idx >= 1024 * 512) return;
    int n = idx >> 9, k = idx & 511;
    WoT[idx] = f2bf(Wo[k * 1024 + n]);         // WoT[n][k] = Wo[k][n]
}

__global__ __launch_bounds__(256) void pack_wh(const float* __restrict__ W,
                                               uint16_t* __restrict__ Wh) {
    int idx = blockIdx.x * 256 + threadIdx.x;  // 8*64*64
    if (idx >= 8 * 64 * 64) return;
    int n = idx >> 12, rem = idx & 4095, w = rem >> 6, v = rem & 63;
    Wh[idx] = f2bf(W[n * 4096 + v * 64 + w]);  // Wh[n][w][v] = W[n][v][w]
}

// ---------------------------------------------------------------------------
// Tiled bf16 WMMA GEMM: C(MxN fp32) = A(MxK fp32) * B, with B given transposed
// as Bt (N x K, bf16).  128x128 tile per 256-thread block, BK=32 per step.
//  - B tile (shared by all 8 waves) double-buffered in LDS via async
//    global->LDS copies (ASYNCcnt); one barrier per K-step.
//  - A rows are private to one wave: fragments loaded straight from global
//    into registers and converted with v_cvt_pk_bf16_f32; prefetched 1 step.
//  - B fragments stream through a 2-deep rotating register pipeline so two
//    ds_load_b128 pairs stay in flight ahead of each WMMA.
// ---------------------------------------------------------------------------
#define BM 128
#define BN 128
#define BK 32
#define LDT 40  // padded LDS row stride (bf16 elems) to spread banks

__global__ __launch_bounds__(256) void gemm_bf16(const float* __restrict__ A,
                                                 const uint16_t* __restrict__ Bt,
                                                 float* __restrict__ C,
                                                 int M, int N, int K) {
    __shared__ uint16_t Bs[2][BN * LDT];
    int tid = threadIdx.x, lane = tid & 31, wave = tid >> 5;
    int half = lane >> 4;
    int nTilesN = N / BN;
    int bm = (blockIdx.x / nTilesN) * BM;
    int bn = (blockIdx.x % nTilesN) * BN;

    v8f acc[8];
#pragma unroll
    for (int i = 0; i < 8; ++i) acc[i] = v8f_zero();

    // B fill mapping (async): thread covers 32B of one B row per K-step
    int frow = tid >> 1;             // 0..127
    int fcol = (tid & 1) * 16;       // 0 or 16
    uint32_t bgbyte = (uint32_t)(((bn + frow) * K + fcol) * 2);

    // per-lane A fragment base: row bm+wave*16+(lane&15), K offset half*8
    const float* ap = A + (size_t)(bm + wave * 16 + (lane & 15)) * K + half * 8;

    // ---- prologue: async-fill Bs[0], preload A frag floats for kb=0 ----
    {
        uint32_t l0 = lds_off(&Bs[0][frow * LDT + fcol]);
        async_copy_b128(l0,       Bt, bgbyte);
        async_copy_b128(l0 + 16u, Bt, bgbyte + 16u);
    }
    float4 f0 = *(const float4*)(ap);
    float4 f1 = *(const float4*)(ap + 4);
    float4 f2 = *(const float4*)(ap + 16);
    float4 f3 = *(const float4*)(ap + 20);
    wait_async0();
    __syncthreads();

    int nk = K / BK;
    for (int i = 0; i < nk; ++i) {
        int p = i & 1;
        bool more = (i + 1) < nk;
        float4 g0, g1, g2, g3;
        if (more) {
            int kb1 = (i + 1) * BK;
            uint32_t l0 = lds_off(&Bs[p ^ 1][frow * LDT + fcol]);
            uint32_t gb = bgbyte + (uint32_t)(kb1 * 2);
            async_copy_b128(l0,       Bt, gb);
            async_copy_b128(l0 + 16u, Bt, gb + 16u);
            g0 = *(const float4*)(ap + kb1);
            g1 = *(const float4*)(ap + kb1 + 4);
            g2 = *(const float4*)(ap + kb1 + 16);
            g3 = *(const float4*)(ap + kb1 + 20);
        }
        // ---- compute from buffer p: 2-deep B-fragment pipeline, 8 WMMAs ----
        v16bf af  = build_fragA(f0, f1, f2, f3);
        v16bf bc  = load_fragB(&Bs[p][0], LDT, 0,  0, lane);
        v16bf bnx = load_fragB(&Bs[p][0], LDT, 16, 0, lane);
#pragma unroll
        for (int nt = 0; nt < 8; ++nt) {
            v16bf bfut;
            if (nt + 2 < 8)
                bfut = load_fragB(&Bs[p][0], LDT, (nt + 2) * 16, 0, lane);
            acc[nt] = __builtin_amdgcn_wmma_f32_16x16x32_bf16(
                false, af, false, bc, (short)0, acc[nt], false, false);
            bc = bnx;
            bnx = bfut;
        }
        if (more) { f0 = g0; f1 = g1; f2 = g2; f3 = g3; }
        wait_async0();
        __syncthreads();   // buffer p^1 now ready; everyone done reading p
    }

    // ---- epilogue: C/D layout -> N = lane&15, M = r + 8*(lane>=16) ----
    int c = lane & 15;
    int mrow = bm + wave * 16 + half * 8;
#pragma unroll
    for (int nt = 0; nt < 8; ++nt) {
        float* cp = C + (size_t)mrow * N + bn + nt * 16 + c;
#pragma unroll
        for (int r = 0; r < 8; ++r) cp[(size_t)r * N] = acc[nt][r];
    }
}

// ---------------------------------------------------------------------------
// Causal depthwise conv (KC=4) + SiLU.
// ---------------------------------------------------------------------------
__global__ __launch_bounds__(256) void conv_silu(const float* __restrict__ in, int ldin,
                                                 const float* __restrict__ cw,
                                                 float* __restrict__ out, int L) {
    int idx = blockIdx.x * 256 + threadIdx.x;  // over B*L*512
    int c  = idx & 511;
    int bl = idx >> 9;
    int l  = bl & (L - 1);
    float4 w = *(const float4*)(cw + c * 4);
    float wv[4] = {w.x, w.y, w.z, w.w};
    float s = 0.0f;
#pragma unroll
    for (int j = 0; j < 4; ++j) {
        int lj = l - 3 + j;
        if (lj >= 0) s += in[(size_t)(bl + lj - l) * ldin + c] * wv[j];
    }
    float sig = 1.0f / (1.0f + __expf(-s));
    out[(size_t)bl * 512 + c] = s * sig;
}

// ---------------------------------------------------------------------------
// f = sigmoid(x @ Wf + bf) : one 256-thread block per (b,l) row.
// ---------------------------------------------------------------------------
__global__ __launch_bounds__(256) void fgate(const float* __restrict__ x,
                                             const float* __restrict__ Wf,
                                             const float* __restrict__ bf,
                                             float* __restrict__ f) {
    __shared__ float part[32][8];
    int row = blockIdx.x;              // b*L + l
    int t = threadIdx.x;
    int n = t & 7, p = t >> 3;
    const float* xr = x + (size_t)row * 1024;
    float s = 0.0f;
#pragma unroll 4
    for (int d = p * 32; d < p * 32 + 32; ++d) s += xr[d] * Wf[d * 8 + n];
    part[p][n] = s;
    __syncthreads();
    if (t < 8) {
        float acc = bf[t];
#pragma unroll
        for (int pp = 0; pp < 32; ++pp) acc += part[pp][t];
        f[(size_t)row * 8 + t] = 1.0f / (1.0f + __expf(-acc));
    }
}

// ---------------------------------------------------------------------------
// Sequential recurrence: one workgroup per (b,n).  H lives in REGISTERS
// (each lane statically owns 16 elements in C/D layout); only the bf16
// shadow needed by the cross-lane WMMA A-fragments goes to LDS.  q/k/v/f
// register-prefetched one step ahead; fast-tanh gate blend; y_t partials
// folded into the register update loop.  3 barriers per step.
// ---------------------------------------------------------------------------
#define HBLD 72  // bf16 row stride (pad 8, keeps 16B alignment)

__global__ __launch_bounds__(256) void recurrence(const float* __restrict__ q,
                                                  const float* __restrict__ k,
                                                  const float* __restrict__ v,
                                                  const float* __restrict__ f,
                                                  const uint16_t* __restrict__ Wh,
                                                  float* __restrict__ y,
                                                  float* __restrict__ Hout, int L) {
    __shared__ uint16_t Hb[64 * HBLD];
    __shared__ uint16_t Wb[64 * HBLD];
    __shared__ float qs[64], ks[64], vs[64];
    __shared__ float fs;
    __shared__ float yp[8][64];

    int b = blockIdx.x >> 3, n = blockIdx.x & 7;
    int tid = threadIdx.x, lane = tid & 31, wave = tid >> 5;
    int half = lane >> 4, cN = lane & 15;

    // per-wave static tiles: tt = wave*2+i; tm=(tt>>2)*16, tn=(tt&3)*16
    float hreg[2][8];
#pragma unroll
    for (int i = 0; i < 2; ++i)
#pragma unroll
        for (int r = 0; r < 8; ++r) hreg[i][r] = 0.0f;

    // init Hb = 0, load per-head W^T (bf16) into LDS
    for (int i = tid; i < 64 * 64; i += 256) {
        int r = i >> 6, c = i & 63;
        Hb[r * HBLD + c] = 0;
        Wb[r * HBLD + c] = Wh[((size_t)n * 64 + r) * 64 + c];
    }

    const float* qb = q + ((size_t)b * L * 8 + n) * 64;
    const float* kb = k + ((size_t)b * L * 8 + n) * 64;
    const float* vb = v + ((size_t)b * L * 8 + n) * 64;
    const float* fb = f + ((size_t)b * L * 8 + n);

    // register preload for t = 0
    float rq = 0.0f, rk = 0.0f, rv = 0.0f, rf = 0.0f;
    if (tid < 64)        rq = qb[tid];
    else if (tid < 128)  rk = kb[tid - 64];
    else if (tid < 192)  rv = vb[tid - 128];
    else if (tid == 192) rf = fb[0];
    __syncthreads();

    for (int t = 0; t < L; ++t) {
        // ---- stage prefetched step-t operands into LDS ----
        if (tid < 64)        qs[tid]       = rq;
        else if (tid < 128)  ks[tid - 64]  = rk;
        else if (tid < 192)  vs[tid - 128] = rv;
        else if (tid == 192) fs = rf;
        // ---- register prefetch for step t+1 (overlaps whole step) ----
        if (t + 1 < L) {
            size_t o = (size_t)(t + 1) * 512;
            if (tid < 64)        rq = qb[o + tid];
            else if (tid < 128)  rk = kb[o + tid - 64];
            else if (tid < 192)  rv = vb[o + tid - 128];
            else if (tid == 192) rf = fb[(size_t)(t + 1) * 8];
        }
        // warm L2 a few steps out (global_prefetch_b8)
        if (t + 8 < L && tid < 6) {
            const float* pf = (tid >> 1) == 0 ? qb : ((tid >> 1) == 1 ? kb : vb);
            __builtin_prefetch(pf + (size_t)(t + 8) * 512 + (tid & 1) * 32, 0, 1);
        }
        __syncthreads();  // B1

        // ---- cand = H @ W + k v^T  (2 tiles per wave, 2 WMMAs per tile) ----
        v8f acc[2];
#pragma unroll
        for (int i = 0; i < 2; ++i) {
            int tt = wave * 2 + i;
            int tm = (tt >> 2) * 16, tn = (tt & 3) * 16;
            int rowbase = tm + half * 8;
            float vvv = vs[tn + cN];
            float4 k0 = *(const float4*)(&ks[rowbase]);
            float4 k1 = *(const float4*)(&ks[rowbase + 4]);
            acc[i][0] = k0.x * vvv; acc[i][1] = k0.y * vvv;
            acc[i][2] = k0.z * vvv; acc[i][3] = k0.w * vvv;
            acc[i][4] = k1.x * vvv; acc[i][5] = k1.y * vvv;
            acc[i][6] = k1.z * vvv; acc[i][7] = k1.w * vvv;
            v16bf a0 = load_fragA(Hb, HBLD, tm, 0,  lane);
            v16bf a1 = load_fragA(Hb, HBLD, tm, 32, lane);
            v16bf b0 = load_fragB(Wb, HBLD, tn, 0,  lane);
            v16bf b1 = load_fragB(Wb, HBLD, tn, 32, lane);
            acc[i] = __builtin_amdgcn_wmma_f32_16x16x32_bf16(
                false, a0, false, b0, (short)0, acc[i], false, false);
            acc[i] = __builtin_amdgcn_wmma_f32_16x16x32_bf16(
                false, a1, false, b1, (short)0, acc[i], false, false);
        }
        __syncthreads();  // B2: all Hb fragment reads complete before update

        // ---- gated update in registers + y partials + bf16 shadow store ----
        float g = fs;
#pragma unroll
        for (int i = 0; i < 2; ++i) {
            int tt = wave * 2 + i;
            int tm = (tt >> 2) * 16, tn = (tt & 3) * 16;
            int rowbase = tm + half * 8;
            int col = tn + cN;
            float4 q0 = *(const float4*)(&qs[rowbase]);
            float4 q1 = *(const float4*)(&qs[rowbase + 4]);
            float qq[8] = {q0.x, q0.y, q0.z, q0.w, q1.x, q1.y, q1.z, q1.w};
            float yacc = 0.0f;
#pragma unroll
            for (int r = 0; r < 8; ++r) {
                float cand = fast_tanh(acc[i][r]);
                float hnew = g * hreg[i][r] + (1.0f - g) * cand;
                hreg[i][r] = hnew;
                Hb[(rowbase + r) * HBLD + col] = f2bf(hnew);
                yacc += qq[r] * hnew;
            }
            yp[(tm >> 4) * 2 + half][col] = yacc;
        }
        __syncthreads();  // B3

        if (tid < 64) {
            float s = yp[0][tid] + yp[1][tid] + yp[2][tid] + yp[3][tid] +
                      yp[4][tid] + yp[5][tid] + yp[6][tid] + yp[7][tid];
            y[(((size_t)b * L + t) * 8 + n) * 64 + tid] = s;
        }
    }

    // ---- write H_final (B,N,K,V) straight from registers ----
#pragma unroll
    for (int i = 0; i < 2; ++i) {
        int tt = wave * 2 + i;
        int tm = (tt >> 2) * 16, tn = (tt & 3) * 16;
        int rowbase = tm + half * 8;
        int col = tn + cN;
#pragma unroll
        for (int r = 0; r < 8; ++r)
            Hout[(((size_t)b * 8 + n) * 64 + rowbase + r) * 64 + col] = hreg[i][r];
    }
}

// ---------------------------------------------------------------------------
// Launch
// ---------------------------------------------------------------------------
extern "C" void kernel_launch(void* const* d_in, const int* in_sizes, int n_in,
                              void* d_out, int out_size, void* d_ws, size_t ws_size,
                              hipStream_t stream) {
    const float* x   = (const float*)d_in[0];
    const float* Wq  = (const float*)d_in[1];
    const float* Wk  = (const float*)d_in[2];
    const float* Wv  = (const float*)d_in[3];
    const float* Wf  = (const float*)d_in[4];
    const float* bfv = (const float*)d_in[5];
    const float* cwq = (const float*)d_in[6];
    const float* cwk = (const float*)d_in[7];
    const float* cwv = (const float*)d_in[8];
    const float* W   = (const float*)d_in[9];
    const float* Wo  = (const float*)d_in[10];

    const int Bz = 4, L = 1024, D = 1024;
    const int ML = Bz * L;            // 4096 rows
    float* out  = (float*)d_out;                       // (B,L,D)
    float* Hout = out + (size_t)ML * D;                // (B,N,K,V)

    char* ws = (char*)d_ws;
    float* qkv_pre = (float*)ws; ws += (size_t)ML * 1536 * 4;
    float* qc = (float*)ws;      ws += (size_t)ML * 512 * 4;
    float* kc = (float*)ws;      ws += (size_t)ML * 512 * 4;
    float* vc = (float*)ws;      ws += (size_t)ML * 512 * 4;
    float* fg = (float*)ws;      ws += (size_t)ML * 8 * 4;
    float* yb = (float*)ws;      ws += (size_t)ML * 512 * 4;
    uint16_t* Wt  = (uint16_t*)ws; ws += (size_t)1536 * 1024 * 2;
    uint16_t* WoT = (uint16_t*)ws; ws += (size_t)1024 * 512 * 2;
    uint16_t* Whb = (uint16_t*)ws; ws += (size_t)8 * 64 * 64 * 2;

    // weight packing (fp32 -> transposed bf16)
    pack_qkv_w<<<(1536 * 1024 + 255) / 256, 256, 0, stream>>>(Wq, Wk, Wv, Wt);
    pack_wo<<<(1024 * 512 + 255) / 256, 256, 0, stream>>>(Wo, WoT);
    pack_wh<<<(8 * 64 * 64 + 255) / 256, 256, 0, stream>>>(W, Whb);

    // fused q/k/v projection GEMM: (4096x1024)x(1024x1536)
    gemm_bf16<<<(ML / BM) * (1536 / BN), 256, 0, stream>>>(x, Wt, qkv_pre, ML, 1536, 1024);

    // causal depthwise conv + SiLU (column slices of qkv_pre)
    conv_silu<<<(ML * 512) / 256, 256, 0, stream>>>(qkv_pre,        1536, cwq, qc, L);
    conv_silu<<<(ML * 512) / 256, 256, 0, stream>>>(qkv_pre + 512,  1536, cwk, kc, L);
    conv_silu<<<(ML * 512) / 256, 256, 0, stream>>>(qkv_pre + 1024, 1536, cwv, vc, L);

    // forget gate
    fgate<<<ML, 256, 0, stream>>>(x, Wf, bfv, fg);

    // sequential WMMA recurrence: one workgroup per (b,n)
    recurrence<<<32, 256, 0, stream>>>(qc, kc, vc, fg, Whb, yb, Hout, L);

    // output GEMM: (4096x512)x(512x1024)
    gemm_bf16<<<(ML / BM) * (1024 / BN), 256, 0, stream>>>(yb, WoT, out, ML, 1024, 512);
}
